// WindowAttention_53979148976450
// MI455X (gfx1250) — matrix operand order repaired and checked by reading, not compiled, not run on hardware
//
#include <hip/hip_runtime.h>
#include <hip/hip_bf16.h>

typedef __attribute__((ext_vector_type(16))) __bf16 bf16x16;
typedef __attribute__((ext_vector_type(8)))  __bf16 bf16x8;
typedef __attribute__((ext_vector_type(8)))  float  f32x8;

#define NTOK   49
#define EMBED  96
#define HEADS  6
#define SCALE  0.25f

// bf16 weight workspace layout (elements)
#define WQOFF 0
#define WKOFF 9216
#define WVOFF 18432
#define WPOFF 27648
#define WTOT  36864     // 73,728 bytes in d_ws

// per-wave LDS layout (bf16 elements)
#define MS 104          // row stride for Q/K/O (96 cols + pad, multiple of 8)
#define VS 72           // row stride for V^T (64 cols + pad)
#define PS 72           // row stride for P (64 cols + pad)
#define Q0 0
#define K0 6656         // 64*104
#define V0 13312        // +64*104
#define P0 20224        // +96*72
#define O0 21376        // +16*72
#define WAVE_LDS 28032  // +64*104  (bf16 elements per wave)
#define WPB 4           // waves (windows) per block

static __device__ __forceinline__ bf16x16 bzero16() {
  bf16x16 z;
#pragma unroll
  for (int i = 0; i < 16; ++i) z[i] = (__bf16)0.0f;
  return z;
}

// A-fragment (16-bit, 16x32): this lane needs p[0..7] and p[16..23] (fp32 -> bf16)
static __device__ __forceinline__ bf16x16 a_frag_f32(const float* __restrict__ p, bool valid) {
  bf16x16 a = bzero16();
  if (valid) {
    float4 u0 = ((const float4*)p)[0];
    float4 u1 = ((const float4*)p)[1];
    float4 w0 = ((const float4*)(p + 16))[0];
    float4 w1 = ((const float4*)(p + 16))[1];
    a[0]=(__bf16)u0.x;  a[1]=(__bf16)u0.y;  a[2]=(__bf16)u0.z;  a[3]=(__bf16)u0.w;
    a[4]=(__bf16)u1.x;  a[5]=(__bf16)u1.y;  a[6]=(__bf16)u1.z;  a[7]=(__bf16)u1.w;
    a[8]=(__bf16)w0.x;  a[9]=(__bf16)w0.y;  a[10]=(__bf16)w0.z; a[11]=(__bf16)w0.w;
    a[12]=(__bf16)w1.x; a[13]=(__bf16)w1.y; a[14]=(__bf16)w1.z; a[15]=(__bf16)w1.w;
  }
  return a;
}

// 16 contiguous bf16 (two b128 loads) — works for LDS or global pointers
static __device__ __forceinline__ bf16x16 frag_contig16(const __bf16* p) {
  bf16x8 lo = *(const bf16x8*)p;
  bf16x8 hi = *(const bf16x8*)(p + 8);
  bf16x16 a;
#pragma unroll
  for (int i = 0; i < 8; ++i) { a[i] = lo[i]; a[8 + i] = hi[i]; }
  return a;
}

// A-fragment from bf16 memory: p[0..7] and p[16..23]
static __device__ __forceinline__ bf16x16 a_frag_bf16(const __bf16* p) {
  bf16x8 lo = *(const bf16x8*)p;
  bf16x8 hi = *(const bf16x8*)(p + 16);
  bf16x16 a;
#pragma unroll
  for (int i = 0; i < 8; ++i) { a[i] = lo[i]; a[8 + i] = hi[i]; }
  return a;
}

// ---- prep: fp32 weights -> bf16 workspace (re-run every launch) ----
__global__ __launch_bounds__(256) void cvt_weights(
    const float* __restrict__ wq, const float* __restrict__ wk,
    const float* __restrict__ wv, const float* __restrict__ wp,
    __bf16* __restrict__ ws)
{
  const int i = blockIdx.x * 256 + threadIdx.x;
  if (i >= WTOT) return;
  float v;
  if      (i < WKOFF) v = wq[i];
  else if (i < WVOFF) v = wk[i - WKOFF];
  else if (i < WPOFF) v = wv[i - WVOFF];
  else                v = wp[i - WPOFF];
  ws[i] = (__bf16)v;
}

__global__ __launch_bounds__(WPB * 32, 1) void swin_win_attn(
    const float* __restrict__ x,    const float* __restrict__ mask,
    const __bf16* __restrict__ wbf,
    const float* __restrict__ bq,   const float* __restrict__ bk,
    const float* __restrict__ bv,   const float* __restrict__ bp,
    const float* __restrict__ btab, const int* __restrict__ ridx,
    float* __restrict__ out, int nwin)
{
  extern __shared__ __align__(16) __bf16 smem[];
  const int lane = threadIdx.x & 31;
  const int wave = threadIdx.x >> 5;
  const int b = blockIdx.x * WPB + wave;
  if (b >= nwin) return;                 // wave-uniform: EXEC stays all-ones for WMMA
  __bf16* lds = smem + wave * WAVE_LDS;

  const int  l15  = lane & 15;
  const bool hi   = lane >= 16;
  const int  hio8  = hi ? 8 : 0;
  const int  hio16 = hi ? 16 : 0;
  const float* xw = x + (size_t)b * (NTOK * EMBED);

  // ---------------- Phase A: Q, K, V projections ----------------
  for (int mt = 0; mt < 4; ++mt) {
    const int  m  = mt * 16 + l15;
    const bool mv = m < NTOK;
    const float* xr = xw + m * EMBED;
    bf16x16 xa0 = a_frag_f32(xr + 0  + hio8, mv);
    bf16x16 xa1 = a_frag_f32(xr + 32 + hio8, mv);
    bf16x16 xa2 = a_frag_f32(xr + 64 + hio8, mv);
    for (int p = 0; p < 3; ++p) {
      const __bf16* W  = wbf + p * 9216;   // wq, wk, wv (bf16, row-major 96x96)
      const float*  bb = (p == 0) ? bq : ((p == 1) ? bk : bv);
      f32x8 acc[6];
#pragma unroll
      for (int dt = 0; dt < 6; ++dt)
#pragma unroll
        for (int i = 0; i < 8; ++i) acc[dt][i] = 0.0f;
#pragma unroll
      for (int kt = 0; kt < 3; ++kt) {
        bf16x16 xa = (kt == 0) ? xa0 : ((kt == 1) ? xa1 : xa2);
#pragma unroll
        for (int dt = 0; dt < 6; ++dt) {
          bf16x16 wb = a_frag_bf16(W + (dt * 16 + l15) * EMBED + kt * 32 + hio8);
          acc[dt] = __builtin_amdgcn_wmma_f32_16x16x32_bf16(
              false, xa, false, wb, (short)0, acc[dt], false, false);
        }
      }
#pragma unroll
      for (int dt = 0; dt < 6; ++dt) {
        const int   col = dt * 16 + l15;
        const float bl  = bb[col];
#pragma unroll
        for (int r = 0; r < 8; ++r) {
          const float v   = acc[dt][r] + bl;
          const int   row = mt * 16 + r + hio8;
          if (p == 2)      lds[V0 + col * VS + row] = (__bf16)v;   // V transposed
          else if (p == 1) lds[K0 + row * MS + col] = (__bf16)v;
          else             lds[Q0 + row * MS + col] = (__bf16)v;
        }
      }
    }
  }

  // ---------------- Phase B: attention per head ----------------
  const int w = b & 63;  // nW == 64
  for (int h = 0; h < 6; ++h) {
    bf16x16 kf[4];
#pragma unroll
    for (int nt = 0; nt < 4; ++nt) {
      if (!hi) kf[nt] = frag_contig16(lds + K0 + (nt * 16 + l15) * MS + h * 16);
      else     kf[nt] = bzero16();     // K-dim 16..31 is zero padding
    }
    bf16x16 vf[2];
#pragma unroll
    for (int kt = 0; kt < 2; ++kt)
      vf[kt] = a_frag_bf16(lds + V0 + (h * 16 + l15) * VS + kt * 32 + hio8);

    for (int mt = 0; mt < 4; ++mt) {
      bf16x16 qa = bzero16();
      {
        if (!hi) qa = frag_contig16(lds + Q0 + (mt * 16 + l15) * MS + h * 16);   // K 16..31 stays zero
      }
      f32x8 s[4];
#pragma unroll
      for (int nt = 0; nt < 4; ++nt) {
        f32x8 z;
#pragma unroll
        for (int i = 0; i < 8; ++i) z[i] = 0.0f;
        s[nt] = __builtin_amdgcn_wmma_f32_16x16x32_bf16(
            false, qa, false, kf[nt], (short)0, z, false, false);
      }
      // scale + rel-pos bias + window mask + row softmax
#pragma unroll
      for (int r = 0; r < 8; ++r) {
        const int n = mt * 16 + r + hio8;
        float mx = -1e30f;
#pragma unroll
        for (int nt = 0; nt < 4; ++nt) {
          const int mc = nt * 16 + l15;
          float v = -1e30f;
          if (n < NTOK && mc < NTOK) {
            const int ri = ridx[n * NTOK + mc];
            v = s[nt][r] * SCALE + btab[ri * HEADS + h]
              + mask[((size_t)w * NTOK + n) * NTOK + mc];
          }
          s[nt][r] = v;
          mx = fmaxf(mx, v);
        }
        mx = fmaxf(mx, __shfl_xor(mx, 8));
        mx = fmaxf(mx, __shfl_xor(mx, 4));
        mx = fmaxf(mx, __shfl_xor(mx, 2));
        mx = fmaxf(mx, __shfl_xor(mx, 1));
        float sum = 0.0f;
#pragma unroll
        for (int nt = 0; nt < 4; ++nt) {
          float e = __expf(s[nt][r] - mx);
          s[nt][r] = e;
          sum += e;
        }
        sum += __shfl_xor(sum, 8);
        sum += __shfl_xor(sum, 4);
        sum += __shfl_xor(sum, 2);
        sum += __shfl_xor(sum, 1);
        const float inv = __builtin_amdgcn_rcpf(sum);
#pragma unroll
        for (int nt = 0; nt < 4; ++nt) s[nt][r] *= inv;
      }
      // P -> LDS (bf16): D-layout -> A-layout transpose via LDS
#pragma unroll
      for (int nt = 0; nt < 4; ++nt)
#pragma unroll
        for (int r = 0; r < 8; ++r)
          lds[P0 + (r + hio8) * PS + nt * 16 + l15] = (__bf16)s[nt][r];
      // O_tile = P @ V_h
      f32x8 o;
#pragma unroll
      for (int i = 0; i < 8; ++i) o[i] = 0.0f;
#pragma unroll
      for (int kt = 0; kt < 2; ++kt) {
        bf16x16 pa = a_frag_bf16(lds + P0 + l15 * PS + kt * 32 + hio8);
        o = __builtin_amdgcn_wmma_f32_16x16x32_bf16(
            false, pa, false, vf[kt], (short)0, o, false, false);
      }
#pragma unroll
      for (int r = 0; r < 8; ++r)
        lds[O0 + (mt * 16 + r + hio8) * MS + h * 16 + l15] = (__bf16)o[r];
    }
  }

  // ---------------- Phase C: output projection ----------------
  const __bf16* Wp = wbf + WPOFF;
  for (int mt = 0; mt < 4; ++mt) {
    const __bf16* orow = lds + O0 + (mt * 16 + l15) * MS;
    bf16x16 oa0 = a_frag_bf16(orow + 0  + hio8);
    bf16x16 oa1 = a_frag_bf16(orow + 32 + hio8);
    bf16x16 oa2 = a_frag_bf16(orow + 64 + hio8);
    f32x8 acc[6];
#pragma unroll
    for (int dt = 0; dt < 6; ++dt)
#pragma unroll
      for (int i = 0; i < 8; ++i) acc[dt][i] = 0.0f;
#pragma unroll
    for (int kt = 0; kt < 3; ++kt) {
      bf16x16 oa = (kt == 0) ? oa0 : ((kt == 1) ? oa1 : oa2);
#pragma unroll
      for (int dt = 0; dt < 6; ++dt) {
        bf16x16 wb = a_frag_bf16(Wp + (dt * 16 + l15) * EMBED + kt * 32 + hio8);
        acc[dt] = __builtin_amdgcn_wmma_f32_16x16x32_bf16(
            false, oa, false, wb, (short)0, acc[dt], false, false);
      }
    }
#pragma unroll
    for (int dt = 0; dt < 6; ++dt) {
      const float bl = bp[dt * 16 + l15];
#pragma unroll
      for (int r = 0; r < 8; ++r) {
        const int n = mt * 16 + r + hio8;
        if (n < NTOK)
          out[((size_t)b * NTOK + n) * EMBED + dt * 16 + l15] = acc[dt][r] + bl;
      }
    }
  }
}

extern "C" void kernel_launch(void* const* d_in, const int* in_sizes, int n_in,
                              void* d_out, int out_size, void* d_ws, size_t ws_size,
                              hipStream_t stream) {
  (void)n_in; (void)out_size; (void)ws_size;
  const float* x    = (const float*)d_in[0];
  const float* mask = (const float*)d_in[1];
  const float* wq   = (const float*)d_in[2];
  const float* bq   = (const float*)d_in[3];
  const float* wk   = (const float*)d_in[4];
  const float* bk   = (const float*)d_in[5];
  const float* wv   = (const float*)d_in[6];
  const float* bv   = (const float*)d_in[7];
  const float* wp   = (const float*)d_in[8];
  const float* bp   = (const float*)d_in[9];
  const float* btab = (const float*)d_in[10];
  const int*   ridx = (const int*)d_in[11];
  float*  out = (float*)d_out;
  __bf16* wbf = (__bf16*)d_ws;

  // prep: fp32 -> bf16 weights in workspace (deterministic, every call)
  cvt_weights<<<(WTOT + 255) / 256, 256, 0, stream>>>(wq, wk, wv, wp, wbf);

  const int nwin   = in_sizes[0] / (NTOK * EMBED);   // 8192
  const int blocks = (nwin + WPB - 1) / WPB;
  const size_t shmem = (size_t)WPB * WAVE_LDS * sizeof(__bf16);  // 224,256 B

  swin_win_attn<<<blocks, WPB * 32, shmem, stream>>>(
      x, mask, wbf, bq, bk, bv, bp, btab, ridx, out, nwin);
}
